// PCCNet_33320356282792
// MI455X (gfx1250) — matrix operand order repaired and checked by reading, not compiled, not run on hardware
//
#include <hip/hip_runtime.h>
#include <hip/hip_bf16.h>

typedef __attribute__((ext_vector_type(16))) _Float16 v16h;
typedef __attribute__((ext_vector_type(8)))  float    v8f;

#define NPTS 4096
#define NBATCH 8
#define KNN_K 30
#define SLOPE 0.2f

// K index inside a 16-bit A-matrix (16x32) tile for element e of the per-lane
// v16h, given lane-half lh (ISA 7.12.2: VGPR v<4 -> K = lh*8 + v*2 + h,
// v>=4 -> K = 16 + lh*8 + (v-4)*2 + h).
__device__ __forceinline__ int a_k_idx(int e, int lh) {
  int v = e >> 1, h = e & 1;
  return ((v & 4) ? 16 : 0) + lh * 8 + (v & 3) * 2 + h;
}

// ---------------------------------------------------------------------------
// Kernel 1: brute-force KNN (top-30 by pd = 2*inner - |q|^2 - |p|^2)
// ---------------------------------------------------------------------------
__global__ __launch_bounds__(256) void knn_kernel(const float* __restrict__ x,
                                                  int* __restrict__ idx_out) {
  const int tilesPerBatch = NPTS / 256;
  int b = blockIdx.x / tilesPerBatch;
  int n = (blockIdx.x % tilesPerBatch) * 256 + threadIdx.x;
  const float* xb = x + b * 3 * NPTS;
  float qx = xb[n], qy = xb[NPTS + n], qz = xb[2 * NPTS + n];
  float qq = qx * qx + qy * qy + qz * qz;

  float bd[KNN_K];
  int   bi[KNN_K];
#pragma unroll
  for (int i = 0; i < KNN_K; ++i) { bd[i] = -3.4e38f; bi[i] = 0; }

  __shared__ float sx[256], sy[256], sz[256];
  for (int t0 = 0; t0 < NPTS; t0 += 256) {
    int m = t0 + threadIdx.x;
    sx[threadIdx.x] = xb[m];
    sy[threadIdx.x] = xb[NPTS + m];
    sz[threadIdx.x] = xb[2 * NPTS + m];
    if (t0 + 256 < NPTS) {
      __builtin_prefetch(&xb[t0 + 256 + threadIdx.x], 0, 0);
      __builtin_prefetch(&xb[NPTS + t0 + 256 + threadIdx.x], 0, 0);
      __builtin_prefetch(&xb[2 * NPTS + t0 + 256 + threadIdx.x], 0, 0);
    }
    __syncthreads();
    for (int mm = 0; mm < 256; ++mm) {
      float px = sx[mm], py = sy[mm], pz = sz[mm];
      float inner = qx * px + qy * py + qz * pz;
      float pp = px * px + py * py + pz * pz;
      float pd = 2.0f * inner - qq - pp;
      if (pd > bd[KNN_K - 1]) {
        bd[KNN_K - 1] = pd;
        bi[KNN_K - 1] = t0 + mm;
#pragma unroll
        for (int s2 = KNN_K - 1; s2 > 0; --s2) {
          if (bd[s2] > bd[s2 - 1]) {
            float td = bd[s2]; bd[s2] = bd[s2 - 1]; bd[s2 - 1] = td;
            int   ti = bi[s2]; bi[s2] = bi[s2 - 1]; bi[s2 - 1] = ti;
          }
        }
      }
    }
    __syncthreads();
  }
  int base = (b * NPTS + n) * KNN_K;
#pragma unroll
  for (int i = 0; i < KNN_K; ++i) idx_out[base + i] = bi[i];
}

// ---------------------------------------------------------------------------
// Kernel 2: fused multi-scale MLP + attention, one wave32 per 16-point tile.
// Layer1: h1(64x16) = W1(64x6, K-padded to 32) x F(32x16)      -> 4 WMMA
// Layer2: h2(64x16) = W2(64x64) x h1(64x16)                    -> 8 WMMA
// ---------------------------------------------------------------------------
__global__ __launch_bounds__(32) void pccnet_kernel(
    const float* __restrict__ x,
    const float* __restrict__ w1, const float* __restrict__ g1, const float* __restrict__ b1,
    const float* __restrict__ w2, const float* __restrict__ g2, const float* __restrict__ b2,
    const float* __restrict__ wa, const float* __restrict__ ba,
    const float* __restrict__ ga, const float* __restrict__ bga,
    const int* __restrict__ knn_idx, float* __restrict__ out) {

  int b    = blockIdx.x >> 8;           // 256 tiles of 16 points per batch
  int n0   = (blockIdx.x & 255) << 4;
  int lane = threadIdx.x;
  int col  = lane & 15;                 // column / point within tile
  int lh   = lane >> 4;                 // lane half
  int mbase = lh * 8;                   // C-layout row offset within 16x16 tile

  __shared__ float s_g1[64], s_b1[64], s_g2[64], s_b2[64], s_wa[64];
  __shared__ __align__(32) _Float16 h1s[16][64];   // h1 re-layout buffer [col][chan]
  __shared__ float a_lds[16][32];                  // attention logits/probs

  const float invs = rsqrtf(1.0f + 1e-5f);
  for (int i = lane; i < 64; i += 32) {
    s_g1[i] = g1[i] * invs; s_b1[i] = b1[i];
    s_g2[i] = g2[i] * invs; s_b2[i] = b2[i];
    s_wa[i] = wa[i];
  }
  float ba0  = ba[0];
  float gasc = ga[0] * invs;
  float bga0 = bga[0];
  __syncthreads();

  const float* xb = x + b * 3 * NPTS;
  float cx = xb[n0 + col], cy = xb[NPTS + n0 + col], cz = xb[2 * NPTS + n0 + col];

  // Preload W1 / W2 into per-lane A-matrix layout (f16).
  v16h a1[4];
  v16h a2m[4][2];
#pragma unroll
  for (int t = 0; t < 4; ++t) {
    int o = t * 16 + col;               // output channel row held by this lane
#pragma unroll
    for (int e = 0; e < 16; ++e) {
      int i = a_k_idx(e, lh);
      a1[t][e] = (i < 6) ? (_Float16)w1[o * 6 + i] : (_Float16)0.0f;
    }
#pragma unroll
    for (int ks = 0; ks < 2; ++ks) {
#pragma unroll
      for (int e = 0; e < 16; ++e) {
        int i = ks * 32 + lh * 16 + e;
        a2m[t][ks][e] = (_Float16)w2[o * 64 + i];
      }
    }
  }

  // Compute post-activation h2 (C layout) for neighbor group j.
  auto compute_h2 = [&](int j, v8f h2t[4]) {
    // F B-tile (32x16): rows 0..5 = [neigh-center ; center], rest zero.
    v16h fB = {};
    if (lh == 0) {
      int ni = knn_idx[(b * NPTS + n0 + col) * KNN_K + j];
      float nx1 = xb[ni], ny1 = xb[NPTS + ni], nz1 = xb[2 * NPTS + ni];
      fB[0] = (_Float16)(nx1 - cx);
      fB[1] = (_Float16)(ny1 - cy);
      fB[2] = (_Float16)(nz1 - cz);
      fB[3] = (_Float16)cx;
      fB[4] = (_Float16)cy;
      fB[5] = (_Float16)cz;
    }
    // Layer 1: 4 WMMAs, then BN1 + LeakyReLU, stage as f16 into LDS.
#pragma unroll
    for (int t = 0; t < 4; ++t) {
      v8f c = {};
      c = __builtin_amdgcn_wmma_f32_16x16x32_f16(false, a1[t], false, fB,
                                                 (short)0, c, false, false);
      int mb = t * 16 + mbase;
#pragma unroll
      for (int r = 0; r < 8; ++r) {
        float v = c[r] * s_g1[mb + r] + s_b1[mb + r];
        v = (v > 0.0f) ? v : SLOPE * v;
        h1s[col][mb + r] = (_Float16)v;
      }
    }
    __syncthreads();
    // Re-read h1 as two B-tiles (K=0..31 and K=32..63), 32B-aligned v16h.
    const v16h* hp = reinterpret_cast<const v16h*>(&h1s[col][0]);
    v16h bt0 = hp[lh];
    v16h bt1 = hp[2 + lh];
    // Layer 2: 8 WMMAs (2 K-steps x 4 row tiles), BN2 + LeakyReLU.
#pragma unroll
    for (int t = 0; t < 4; ++t) {
      v8f c = {};
      c = __builtin_amdgcn_wmma_f32_16x16x32_f16(false, a2m[t][0], false, bt0,
                                                 (short)0, c, false, false);
      c = __builtin_amdgcn_wmma_f32_16x16x32_f16(false, a2m[t][1], false, bt1,
                                                 (short)0, c, false, false);
      int mb = t * 16 + mbase;
#pragma unroll
      for (int r = 0; r < 8; ++r) {
        float v = c[r] * s_g2[mb + r] + s_b2[mb + r];
        c[r] = (v > 0.0f) ? v : SLOPE * v;
      }
      h2t[t] = c;
    }
    __syncthreads();
  };

  v8f accT[4] = {};   // running sum over scales, C layout

  const int MS[3] = {10, 20, 30};
  for (int s = 0; s < 3; ++s) {
    int kk = MS[s];
    // Pass 1: attention logits.
    for (int j = 0; j < kk; ++j) {
      v8f h2t[4];
      compute_h2(j, h2t);
      float part = 0.0f;
#pragma unroll
      for (int t = 0; t < 4; ++t) {
        int mb = t * 16 + mbase;
#pragma unroll
        for (int r = 0; r < 8; ++r) part += s_wa[mb + r] * h2t[t][r];
      }
      part += __shfl_xor(part, 16, 32);       // fold lane halves (wave32)
      if (lh == 0) {
        float sA = (part + ba0) * gasc + bga0;
        sA = (sA > 0.0f) ? sA : SLOPE * sA;
        a_lds[col][j] = sA;
      }
    }
    __syncthreads();
    // Softmax over k per point (lanes 0..15, one point each).
    if (lh == 0) {
      float mx = -3.4e38f;
      for (int j = 0; j < kk; ++j) mx = fmaxf(mx, a_lds[col][j]);
      float sum = 0.0f;
      for (int j = 0; j < kk; ++j) {
        float e = __expf(a_lds[col][j] - mx);
        a_lds[col][j] = e;
        sum += e;
      }
      float rs = 1.0f / sum;
      for (int j = 0; j < kk; ++j) a_lds[col][j] *= rs;
    }
    __syncthreads();
    // Pass 2: recompute h2 and accumulate weighted sum.
    for (int j = 0; j < kk; ++j) {
      v8f h2t[4];
      compute_h2(j, h2t);
      float p = a_lds[col][j];
#pragma unroll
      for (int t = 0; t < 4; ++t) {
#pragma unroll
        for (int r = 0; r < 8; ++r) accT[t][r] += p * h2t[t][r];
      }
    }
    __syncthreads();
  }

  const float inv3 = 1.0f / 3.0f;
#pragma unroll
  for (int t = 0; t < 4; ++t) {
    int mb = t * 16 + mbase;
#pragma unroll
    for (int r = 0; r < 8; ++r) {
      out[(b * 64 + mb + r) * NPTS + n0 + col] = accT[t][r] * inv3;
    }
  }
}

// ---------------------------------------------------------------------------
extern "C" void kernel_launch(void* const* d_in, const int* in_sizes, int n_in,
                              void* d_out, int out_size, void* d_ws, size_t ws_size,
                              hipStream_t stream) {
  (void)in_sizes; (void)n_in; (void)out_size; (void)ws_size;
  const float* x   = (const float*)d_in[0];
  const float* w1  = (const float*)d_in[1];
  const float* g1  = (const float*)d_in[2];
  const float* b1  = (const float*)d_in[3];
  const float* w2  = (const float*)d_in[4];
  const float* g2  = (const float*)d_in[5];
  const float* b2  = (const float*)d_in[6];
  const float* wa  = (const float*)d_in[7];
  const float* ba  = (const float*)d_in[8];
  const float* ga  = (const float*)d_in[9];
  const float* bga = (const float*)d_in[10];

  int* idxws = (int*)d_ws;   // B*N*30 ints = 3.93 MB of scratch

  knn_kernel<<<dim3(NBATCH * (NPTS / 256)), dim3(256), 0, stream>>>(x, idxws);
  pccnet_kernel<<<dim3(NBATCH * (NPTS / 16)), dim3(32), 0, stream>>>(
      x, w1, g1, b1, w2, g2, b2, wa, ba, ga, bga, idxws, (float*)d_out);
}
